// Mlp_45449343926805
// MI455X (gfx1250) — 8-bit matrix operand order repaired and checked by reading, compiled, not run on hardware
//
#include <hip/hip_runtime.h>
#include <math.h>


typedef __attribute__((ext_vector_type(8))) int v8i;


__device__ __forceinline__ float clampq(float v) {
    return fminf(fmaxf(v, -128.0f), 127.0f);
}

__device__ __forceinline__ unsigned pack4_q(float a, float b, float c, float d) {
    int q0 = (int)rintf(a), q1 = (int)rintf(b), q2 = (int)rintf(c), q3 = (int)rintf(d);
    return (unsigned)(q0 & 255) | ((unsigned)(q1 & 255) << 8) |
           ((unsigned)(q2 & 255) << 16) | ((unsigned)(q3 & 255) << 24);
}

__device__ __forceinline__ void async_copy32(unsigned lds_addr, const uint8_t* gaddr) {
    asm volatile("global_load_async_to_lds_b128 %0, %1, off\n\t"
                 "global_load_async_to_lds_b128 %0, %1, off offset:16"
                 :: "v"(lds_addr), "v"(gaddr)
                 : "memory");
}

__device__ __forceinline__ void wait_async0() {
    asm volatile("s_wait_asynccnt 0x0" ::: "memory");
}

__global__ void k_init(unsigned* scr) {
    if (threadIdx.x < 8) scr[threadIdx.x] = 0u;
}

__global__ void k_absmax(const float* __restrict__ p, size_t n,
                         const float* __restrict__ presc, unsigned* out) {
    float m = 0.0f;
    for (size_t i = (size_t)blockIdx.x * blockDim.x + threadIdx.x; i < n;
         i += (size_t)gridDim.x * blockDim.x)
        m = fmaxf(m, fabsf(p[i]));
    if (presc) m *= fabsf(presc[0]);
    __shared__ float red[256];
    red[threadIdx.x] = m;
    __syncthreads();
    for (int o = 128; o > 0; o >>= 1) {
        if (threadIdx.x < (unsigned)o)
            red[threadIdx.x] = fmaxf(red[threadIdx.x], red[threadIdx.x + o]);
        __syncthreads();
    }
    if (threadIdx.x == 0) atomicMax(out, __float_as_uint(red[0]));
}

__global__ void k_quant_A(const float* __restrict__ src, int Kdim, int Mtiles,
                          const unsigned* __restrict__ maxbits,
                          const float* __restrict__ presc,
                          unsigned* __restrict__ dst) {
    const int tile = blockIdx.x;
    const int kt = tile / Mtiles, mt = tile - kt * Mtiles;
    const int t = threadIdx.x;
    const int lane = t >> 3, v = t & 7;
    const int m = mt * 16 + (lane & 15);
    const int k = kt * 64 + ((v >> 1) << 4) + ((lane >> 4) << 3) + ((v & 1) << 2);
    const float mx = __uint_as_float(maxbits[0]);
    const float ps = presc ? presc[0] : 1.0f;
    const float sc = (mx > 0.0f) ? (127.0f / mx) * ps : 0.0f;
    const float4 f = *(const float4*)(src + (size_t)m * Kdim + k);
    dst[(size_t)tile * 256 + t] =
        pack4_q(clampq(f.x * sc), clampq(f.y * sc), clampq(f.z * sc), clampq(f.w * sc));
}

__global__ void k_quant_B(const float* __restrict__ w, int Kdim, int Ntiles,
                          const unsigned* __restrict__ maxbits,
                          unsigned* __restrict__ dst) {
    const int tile = blockIdx.x;
    const int kt = tile / Ntiles, nt = tile - kt * Ntiles;
    const int t = threadIdx.x;
    const int lane = t >> 3, v = t & 7;
    const int n = nt * 16 + (lane & 15);
    const int k = kt * 64 + ((v >> 1) << 4) + ((lane >> 4) << 3) + ((v & 1) << 2);
    const float mx = __uint_as_float(maxbits[0]);
    const float sc = (mx > 0.0f) ? (127.0f / mx) : 0.0f;
    const float4 f = *(const float4*)(w + (size_t)n * Kdim + k);
    dst[(size_t)tile * 256 + t] =
        pack4_q(clampq(f.x * sc), clampq(f.y * sc), clampq(f.z * sc), clampq(f.w * sc));
}

union Frag32 { uint4 q[2]; v8i v; };

__device__ __forceinline__ void gemm_tiles_iu8(const uint8_t* __restrict__ Asw,
                                               const uint8_t* __restrict__ Bsw,
                                               int Mtiles, int Ntiles, int Ktiles,
                                               v8i acc[4][2]) {
    __shared__ uint8_t ldsA[2][8192];
    __shared__ uint8_t ldsB[2][8192];
    const int tid = threadIdx.x;
    const int lane = tid & 31;
    const int wave = tid >> 5;
    const int waveM = wave >> 2;
    const int waveN = wave & 3;
    const int mBlock = blockIdx.y, nBlock = blockIdx.x;

    const unsigned ldsA_addr = (unsigned)(uintptr_t)(&ldsA[0][0]) + tid * 32;
    const unsigned ldsB_addr = (unsigned)(uintptr_t)(&ldsB[0][0]) + tid * 32;

#pragma unroll
    for (int mi = 0; mi < 4; ++mi)
#pragma unroll
        for (int ni = 0; ni < 2; ++ni)
#pragma unroll
            for (int r = 0; r < 8; ++r) acc[mi][ni][r] = 0;

    async_copy32(ldsA_addr, Asw + (size_t)mBlock * 8 * 1024 + tid * 32);
    async_copy32(ldsB_addr, Bsw + (size_t)nBlock * 8 * 1024 + tid * 32);
    wait_async0();
    __syncthreads();

    for (int kt = 0; kt < Ktiles; ++kt) {
        const int cur = kt & 1;
        const bool more = (kt + 1) < Ktiles;
        if (more) {
            async_copy32(ldsA_addr + (cur ^ 1) * 8192,
                         Asw + ((size_t)(kt + 1) * Mtiles + (size_t)mBlock * 8) * 1024 + tid * 32);
            async_copy32(ldsB_addr + (cur ^ 1) * 8192,
                         Bsw + ((size_t)(kt + 1) * Ntiles + (size_t)nBlock * 8) * 1024 + tid * 32);
            if (kt + 2 < Ktiles) {
                __builtin_prefetch(Asw + ((size_t)(kt + 2) * Mtiles + (size_t)mBlock * 8) * 1024 + tid * 32, 0, 1);
                __builtin_prefetch(Bsw + ((size_t)(kt + 2) * Ntiles + (size_t)nBlock * 8) * 1024 + tid * 32, 0, 1);
            }
        }

        Frag32 af[4], bf[2];
        const uint8_t* LA = ldsA[cur] + waveM * 4096 + lane * 32;
        const uint8_t* LB = ldsB[cur] + waveN * 2048 + lane * 32;
#pragma unroll
        for (int mi = 0; mi < 4; ++mi) {
            af[mi].q[0] = *(const uint4*)(LA + mi * 1024);
            af[mi].q[1] = *(const uint4*)(LA + mi * 1024 + 16);
        }
#pragma unroll
        for (int ni = 0; ni < 2; ++ni) {
            bf[ni].q[0] = *(const uint4*)(LB + ni * 1024);
            bf[ni].q[1] = *(const uint4*)(LB + ni * 1024 + 16);
        }
#pragma unroll
        for (int mi = 0; mi < 4; ++mi)
#pragma unroll
            for (int ni = 0; ni < 2; ++ni)
                acc[mi][ni] = __builtin_amdgcn_wmma_i32_16x16x64_iu8(
                     true, af[mi].v,  true, bf[ni].v,
                    acc[mi][ni],  false,  false);

        if (more) wait_async0();
        __syncthreads();
    }
}

__global__ void __launch_bounds__(256)
k_gemm1(const uint8_t* __restrict__ Asw, const uint8_t* __restrict__ Bsw,
        const float* __restrict__ bias, unsigned* scr,
        float* __restrict__ hout, int Mtiles, int Ntiles, int Ktiles, int Nld) {
    v8i acc[4][2];
    gemm_tiles_iu8(Asw, Bsw, Mtiles, Ntiles, Ktiles, acc);

    const int tid = threadIdx.x, lane = tid & 31, wave = tid >> 5;
    const int waveM = wave >> 2, waveN = wave & 3;
    const float sAB = (__uint_as_float(scr[0]) * (1.0f / 127.0f)) *
                      (__uint_as_float(scr[1]) * (1.0f / 127.0f));
    const int mBase = blockIdx.y * 128 + waveM * 64;
    const int nBase = blockIdx.x * 128 + waveN * 32;
    float lmax = 0.0f;
#pragma unroll
    for (int mi = 0; mi < 4; ++mi)
#pragma unroll
        for (int ni = 0; ni < 2; ++ni) {
            const int row0 = mBase + mi * 16 + (lane >> 4) * 8;
            const int col = nBase + ni * 16 + (lane & 15);
            const float bb = bias[col];
#pragma unroll
            for (int r = 0; r < 8; ++r) {
                float vv = ((float)acc[mi][ni][r] + bb) * sAB;
                float g = 0.5f * vv * (1.0f + erff(vv * 0.70710678118654752f));
                hout[(size_t)(row0 + r) * Nld + col] = g;
                lmax = fmaxf(lmax, fabsf(g));
            }
        }
    __shared__ float red[256];
    red[tid] = lmax;
    __syncthreads();
    for (int o = 128; o > 0; o >>= 1) {
        if (tid < o) red[tid] = fmaxf(red[tid], red[tid + o]);
        __syncthreads();
    }
    if (tid == 0) atomicMax(scr + 3, __float_as_uint(red[0]));
}

__global__ void __launch_bounds__(256)
k_gemm2(const uint8_t* __restrict__ Asw, const uint8_t* __restrict__ Bsw,
        const float* __restrict__ bias, unsigned* scr,
        float* __restrict__ out, int Mtiles, int Ntiles, int Ktiles, int Nld) {
    v8i acc[4][2];
    gemm_tiles_iu8(Asw, Bsw, Mtiles, Ntiles, Ktiles, acc);

    const int tid = threadIdx.x, lane = tid & 31, wave = tid >> 5;
    const int waveM = wave >> 2, waveN = wave & 3;
    const float sAB = (__uint_as_float(scr[3]) * (1.0f / 127.0f)) *
                      (__uint_as_float(scr[2]) * (1.0f / 127.0f));
    const int mBase = blockIdx.y * 128 + waveM * 64;
    const int nBase = blockIdx.x * 128 + waveN * 32;
#pragma unroll
    for (int mi = 0; mi < 4; ++mi)
#pragma unroll
        for (int ni = 0; ni < 2; ++ni) {
            const int row0 = mBase + mi * 16 + (lane >> 4) * 8;
            const int col = nBase + ni * 16 + (lane & 15);
            const float bb = bias[col];
#pragma unroll
            for (int r = 0; r < 8; ++r)
                out[(size_t)(row0 + r) * Nld + col] = ((float)acc[mi][ni][r] + bb) * sAB;
        }
}

extern "C" void kernel_launch(void* const* d_in, const int* in_sizes, int n_in,
                              void* d_out, int out_size, void* d_ws, size_t ws_size,
                              hipStream_t stream) {
    const float* x   = (const float*)d_in[0];
    const float* act = (const float*)d_in[1];
    const float* w1  = (const float*)d_in[2];
    const float* b1  = (const float*)d_in[3];
    const float* w2  = (const float*)d_in[4];
    const float* b2  = (const float*)d_in[5];
    float* out = (float*)d_out;

    const int D = 384, Hd = 1536;
    const int M = in_sizes[0] / D;
    const int MT = M / 16;
    const int KT1 = D / 64, NT1 = Hd / 16;
    const int KT2 = Hd / 64, NT2 = D / 16;

    uint8_t* ws = (uint8_t*)d_ws;
    unsigned* scr = (unsigned*)ws;
    size_t off = 256;
    uint8_t* qA1 = ws + off; off += (size_t)KT1 * MT * 1024;  off = (off + 255) & ~(size_t)255;
    uint8_t* qB1 = ws + off; off += (size_t)KT1 * NT1 * 1024; off = (off + 255) & ~(size_t)255;
    uint8_t* qB2 = ws + off; off += (size_t)KT2 * NT2 * 1024; off = (off + 255) & ~(size_t)255;
    uint8_t* qA2 = ws + off; off += (size_t)KT2 * MT * 1024;  off = (off + 255) & ~(size_t)255;
    float* hbuf  = (float*)(ws + off);

    k_init<<<1, 32, 0, stream>>>(scr);
    k_absmax<<<2048, 256, 0, stream>>>(x,  (size_t)M * D,  act,     scr + 0);
    k_absmax<<<512,  256, 0, stream>>>(w1, (size_t)Hd * D, nullptr, scr + 1);
    k_absmax<<<512,  256, 0, stream>>>(w2, (size_t)D * Hd, nullptr, scr + 2);

    k_quant_A<<<KT1 * MT,  256, 0, stream>>>(x,  D,  MT,  scr + 0, act, (unsigned*)qA1);
    k_quant_B<<<KT1 * NT1, 256, 0, stream>>>(w1, D,  NT1, scr + 1, (unsigned*)qB1);
    k_quant_B<<<KT2 * NT2, 256, 0, stream>>>(w2, Hd, NT2, scr + 2, (unsigned*)qB2);

    dim3 g1(NT1 / 8, M / 128);
    k_gemm1<<<g1, 256, 0, stream>>>(qA1, qB1, b1, scr, hbuf, MT, NT1, KT1, Hd);

    k_quant_A<<<KT2 * MT, 256, 0, stream>>>(hbuf, Hd, MT, scr + 3, nullptr, (unsigned*)qA2);

    dim3 g2(NT2 / 8, M / 128);
    k_gemm2<<<g2, 256, 0, stream>>>(qA2, qB2, b2, scr, out, MT, NT2, KT2, D);
}
